// SequenceLayerGLURope_15324443312610
// MI455X (gfx1250) — matrix operand order repaired and checked by reading, not compiled, not run on hardware
//
#include <hip/hip_runtime.h>
#include <math.h>

// ---------------------------------------------------------------------------
// SequenceLayerGLURope for MI455X (gfx1250), wave32 + WMMA f32_16x16x32_f16.
// All GEMM-shaped work (QKV/coh/GLU/out projections, gb = gk@K^T, per-head
// QK^T and P@V) runs through one generic WMMA GEMM kernel (nn / nt variants).
// Block tile 128x128, 8 waves (4M x 2N), wave tile 32x64 = 2x4 WMMA accs.
// ---------------------------------------------------------------------------

typedef __attribute__((ext_vector_type(16))) _Float16 v16h;
typedef __attribute__((ext_vector_type(8)))  _Float16 v8h;
typedef __attribute__((ext_vector_type(8)))  float    v8f;
typedef __attribute__((ext_vector_type(4)))  float    v4f;

static constexpr int  BB  = 2;
static constexpr int  LL  = 2048;
static constexpr int  DD  = 1024;
static constexpr int  NHH = 16;
static constexpr int  HDD = 64;     // DD / NHH
static constexpr int  HGG = 256;    // rope-gate hidden
static constexpr int  HNN = 128;    // viscosity hidden
static constexpr float BASE_FAST = 1.6180339887498949f;
static constexpr float BASE_SLOW = 1618.0f;
static constexpr float BASE_GLU  = 16180.0f;
static constexpr float ALPHA_ATTN = 0.42f;
static constexpr float ALPHA_MLP  = 1.07f;
static constexpr float LN_EPS = 1e-5f;

// ---------------------------------------------------------------------------
// C = alpha * [tanh(*scale_ptr)] * (A @ op(B)) + bias[col] + add2d[row,col]
// op(B) = B   (transB=0, B is K x N row-major, row stride ldb)
//       = B^T (transB=1, B is N x K row-major, row stride ldb)
// Requirements exploited: K % 32 == 0, N % 16 == 0, 16B-aligned rows.
// ---------------------------------------------------------------------------
__global__ __launch_bounds__(256) void wmma_gemm_kernel(
    const float* __restrict__ A,  int lda,  long strideAb,
    const float* __restrict__ Bm, int ldb,  long strideBb,
    float* __restrict__ C,        int ldc,  long strideCb,
    const float* __restrict__ bias,
    const float* __restrict__ add2d, int ldadd, long strideAddb,
    const float* __restrict__ scale_ptr,
    float alpha, int Mr, int Nr, int Kr, int transB)
{
  __shared__ alignas(16) _Float16 sA[128][32];
  __shared__ alignas(16) _Float16 sB[128][32];

  const int bz = blockIdx.z;
  A  += (long)bz * strideAb;
  Bm += (long)bz * strideBb;
  C  += (long)bz * strideCb;
  if (add2d) add2d += (long)bz * strideAddb;

  const int n0   = blockIdx.x * 128;
  const int m0   = blockIdx.y * 128;
  const int tid  = threadIdx.x;
  const int lane = tid & 31;          // wave32
  const int wave = tid >> 5;          // 0..7
  const int wm   = wave >> 1;         // 4 waves along M
  const int wn   = wave & 1;          // 2 waves along N
  const int half = lane >> 4;         // lane group 0/1
  const int lm   = lane & 15;

  // staging coordinates: each thread stages 16 contiguous values
  const int srow = tid >> 1;          // 0..127 (A row / nt-B row)
  const int skg  = (tid & 1) << 4;    // 0 or 16
  const int bk   = tid >> 3;          // 0..31  (nn-B k)
  const int bng  = (tid & 7) << 4;    // 0..112 (nn-B n group)

  float ascale = alpha;
  if (scale_ptr) ascale *= tanhf(scale_ptr[0]);

  v8f acc[2][4] = {};

  for (int k0 = 0; k0 < Kr; k0 += 32) {
    // ---- stage A tile: sA[m][k] = (f16) A[m0+m][k0+k] ----
    {
      v8h h0 = (v8h)(_Float16)0.f, h1 = (v8h)(_Float16)0.f;
      if (m0 + srow < Mr) {
        const float* src = A + (long)(m0 + srow) * lda + k0 + skg;
        if (k0 + 32 < Kr) __builtin_prefetch(src + 32, 0, 1);
        const v4f f0 = *(const v4f*)(src);
        const v4f f1 = *(const v4f*)(src + 4);
        const v4f f2 = *(const v4f*)(src + 8);
        const v4f f3 = *(const v4f*)(src + 12);
#pragma unroll
        for (int t = 0; t < 4; ++t) {
          h0[t] = (_Float16)f0[t]; h0[4 + t] = (_Float16)f1[t];
          h1[t] = (_Float16)f2[t]; h1[4 + t] = (_Float16)f3[t];
        }
      }
      *(v8h*)&sA[srow][skg]     = h0;   // ds_store_b128
      *(v8h*)&sA[srow][skg + 8] = h1;
    }
    // ---- stage B tile (N-major in LDS): sB[n][k] = (f16) op(B)[k0+k][n0+n] ----
    if (transB) {
      v8h h0 = (v8h)(_Float16)0.f, h1 = (v8h)(_Float16)0.f;
      if (n0 + srow < Nr) {
        const float* src = Bm + (long)(n0 + srow) * ldb + k0 + skg;
        const v4f f0 = *(const v4f*)(src);
        const v4f f1 = *(const v4f*)(src + 4);
        const v4f f2 = *(const v4f*)(src + 8);
        const v4f f3 = *(const v4f*)(src + 12);
#pragma unroll
        for (int t = 0; t < 4; ++t) {
          h0[t] = (_Float16)f0[t]; h0[4 + t] = (_Float16)f1[t];
          h1[t] = (_Float16)f2[t]; h1[4 + t] = (_Float16)f3[t];
        }
      }
      *(v8h*)&sB[srow][skg]     = h0;
      *(v8h*)&sB[srow][skg + 8] = h1;
    } else {
      // transpose through LDS; N % 16 == 0 makes the 16-group guard uniform
      if (n0 + bng < Nr) {
        const float* src = Bm + (long)(k0 + bk) * ldb + n0 + bng;
        const v4f f0 = *(const v4f*)(src);
        const v4f f1 = *(const v4f*)(src + 4);
        const v4f f2 = *(const v4f*)(src + 8);
        const v4f f3 = *(const v4f*)(src + 12);
#pragma unroll
        for (int t = 0; t < 4; ++t) {
          sB[bng +      t][bk] = (_Float16)f0[t];
          sB[bng +  4 + t][bk] = (_Float16)f1[t];
          sB[bng +  8 + t][bk] = (_Float16)f2[t];
          sB[bng + 12 + t][bk] = (_Float16)f3[t];
        }
      } else {
#pragma unroll
        for (int t = 0; t < 16; ++t) sB[bng + t][bk] = (_Float16)0.f;
      }
    }
    __syncthreads();

    // ---- fragments per ISA 7.12.2 16-bit layouts ----
    v16h afrag[2], bfrag[4];
#pragma unroll
    for (int i = 0; i < 2; ++i) {
      const int row = wm * 32 + i * 16 + lm;          // M = lane%16
      const v8h lo = *(const v8h*)&sA[row][half * 8];       // K = half*8 + 0..7
      const v8h hi = *(const v8h*)&sA[row][half * 8 + 16];  // K = half*8 + 16..23
#pragma unroll
      for (int t = 0; t < 8; ++t) { afrag[i][t] = lo[t]; afrag[i][8 + t] = hi[t]; }
    }
#pragma unroll
    for (int j = 0; j < 4; ++j) {
      const int col = wn * 64 + j * 16 + lm;          // N = lane%16
      const v8h lo = *(const v8h*)&sB[col][half * 8];       // K = half*8 + 0..7
      const v8h hi = *(const v8h*)&sB[col][half * 8 + 16];  // K = half*8 + 16..23
#pragma unroll
      for (int t = 0; t < 8; ++t) { bfrag[j][t] = lo[t]; bfrag[j][8 + t] = hi[t]; }
    }

#pragma unroll
    for (int i = 0; i < 2; ++i)
#pragma unroll
      for (int j = 0; j < 4; ++j)
        acc[i][j] = __builtin_amdgcn_wmma_f32_16x16x32_f16(
            /*neg_a=*/false, afrag[i], /*neg_b=*/false, bfrag[j],
            /*c_mod=*/(short)0, acc[i][j], /*reuse_a=*/false, /*reuse_b=*/false);
    __syncthreads();
  }

  // ---- epilogue: C/D layout (lane group selects M-half, lane%16 = N) ----
#pragma unroll
  for (int i = 0; i < 2; ++i)
#pragma unroll
    for (int j = 0; j < 4; ++j) {
      const int col = n0 + wn * 64 + j * 16 + lm;
      if (col >= Nr) continue;
      const float cb = bias ? bias[col] : 0.f;
#pragma unroll
      for (int r = 0; r < 8; ++r) {
        const int row = m0 + wm * 32 + i * 16 + half * 8 + r;
        if (row >= Mr) continue;
        float v = acc[i][j][r] * ascale + cb;
        if (add2d) v += add2d[(long)row * ldadd + col];
        C[(long)row * ldc + col] = v;
      }
    }
}

// ---------------------------------------------------------------------------
// Causal cumulative-mean pool: pool[b,l,d] = mean(z[b,0..l,d])
// ---------------------------------------------------------------------------
__global__ void pool_kernel(const float* __restrict__ z, float* __restrict__ pool)
{
  const int idx = blockIdx.x * blockDim.x + threadIdx.x;   // over B*D
  if (idx >= BB * DD) return;
  const int b = idx / DD, d = idx % DD;
  const float* zp = z + (long)b * LL * DD + d;
  float* pp = pool + (long)b * LL * DD + d;
  float s = 0.f;
  for (int l = 0; l < LL; ++l) {
    s += zp[(long)l * DD];
    pp[(long)l * DD] = s / (float)(l + 1);
  }
}

// ---------------------------------------------------------------------------
// gate (sigmoid(gelu(pool@rg_w1+b)@rg_w2+b)) and nu_b per (b,l).
// Block = one (b,l) row, 256 threads; thread t handles hidden unit t.
// ---------------------------------------------------------------------------
__global__ __launch_bounds__(256) void gate_nu_kernel(
    const float* __restrict__ pool,
    const float* __restrict__ rg_w1, const float* __restrict__ rg_b1,
    const float* __restrict__ rg_w2, const float* __restrict__ rg_b2,
    const float* __restrict__ nu_w1, const float* __restrict__ nu_b1,
    const float* __restrict__ nu_w2, const float* __restrict__ nu_b2,
    const float* __restrict__ nu_diff, const float* __restrict__ nu_adv,
    float* __restrict__ gate, float* __restrict__ nub)
{
  __shared__ float red[256];
  const int bl = blockIdx.x;
  const int t  = threadIdx.x;
  const float* p = pool + (long)bl * DD;

  // rope-gate MLP (HGG == 256 == blockDim)
  float h = rg_b1[t];
  for (int d = 0; d < DD; ++d) h += p[d] * rg_w1[d * HGG + t];
  const float g = 0.5f * h * (1.f + erff(h * 0.70710678118654752f)); // exact gelu
  red[t] = g * rg_w2[t];
  __syncthreads();
  for (int s = 128; s > 0; s >>= 1) { if (t < s) red[t] += red[t + s]; __syncthreads(); }
  if (t == 0) gate[bl] = 1.f / (1.f + expf(-(red[0] + rg_b2[0])));
  __syncthreads();

  // viscosity MLP (HNN == 128)
  float c2 = 0.f;
  if (t < HNN) {
    float hn = nu_b1[t];
    for (int d = 0; d < DD; ++d) hn += p[d] * nu_w1[d * HNN + t];
    c2 = tanhf(hn) * nu_w2[t];
  }
  red[t] = c2;
  __syncthreads();
  for (int s = 128; s > 0; s >>= 1) { if (t < s) red[t] += red[t + s]; __syncthreads(); }
  if (t == 0)
    nub[bl] = fabsf(nu_diff[0]) + tanhf(red[0] + nu_b2[0]) * fabsf(nu_adv[0]);
}

// ---------------------------------------------------------------------------
// RoPE: gated fast/slow mix on Q and K_raw, fixed GLU base on z.
// ---------------------------------------------------------------------------
__global__ void rope_kernel(const float* __restrict__ Q, const float* __restrict__ K,
                            const float* __restrict__ z, const float* __restrict__ gate,
                            float* __restrict__ Qr, float* __restrict__ Kr,
                            float* __restrict__ zglu)
{
  const long idx = (long)blockIdx.x * blockDim.x + threadIdx.x;  // over B*L*D
  if (idx >= (long)BB * LL * DD) return;
  const int  d  = (int)(idx % DD);
  const long bl = idx / DD;
  const int  l  = (int)(bl % LL);
  const int  hi = d & (DD / 2 - 1);                 // d % (D/2)
  const float e = (float)hi / (float)(DD / 2);
  const float t = (float)l;
  const float ff = t * powf(BASE_FAST, -e);
  const float fs = t * powf(BASE_SLOW, -e);
  const float fg = t * powf(BASE_GLU,  -e);
  const float g  = gate[bl];
  const float cm = g * cosf(ff) + (1.f - g) * cosf(fs);
  const float sm = g * sinf(ff) + (1.f - g) * sinf(fs);
  const long base = bl * DD;
  float rq, rk, rz;
  if (d < DD / 2) { rq = -Q[base + d + DD/2]; rk = -K[base + d + DD/2]; rz = -z[base + d + DD/2]; }
  else            { rq =  Q[base + d - DD/2]; rk =  K[base + d - DD/2]; rz =  z[base + d - DD/2]; }
  Qr[base + d]   = Q[base + d] * cm + rq * sm;
  Kr[base + d]   = K[base + d] * cm + rk * sm;
  zglu[base + d] = z[base + d] * cosf(fg) + rz * sinf(fg);
}

// ---------------------------------------------------------------------------
// adv = -nu_b * U * G ; layernorm(adv) * ln_w + ln_b   (block per (b,l) row)
// ---------------------------------------------------------------------------
__global__ __launch_bounds__(256) void glu_ln_kernel(
    const float* __restrict__ U, const float* __restrict__ G,
    const float* __restrict__ nub, const float* __restrict__ ln_w,
    const float* __restrict__ ln_b, float* __restrict__ advn)
{
  __shared__ float red[256];
  const int bl = blockIdx.x;
  const int t  = threadIdx.x;
  const long base = (long)bl * DD;
  const float nu = nub[bl];

  float loc[DD / 256];
  float s = 0.f;
#pragma unroll
  for (int i = 0; i < DD / 256; ++i) {
    const int d = t + i * 256;
    const float a = -nu * U[base + d] * G[base + d];
    loc[i] = a; s += a;
  }
  red[t] = s; __syncthreads();
  for (int k = 128; k > 0; k >>= 1) { if (t < k) red[t] += red[t + k]; __syncthreads(); }
  const float mu = red[0] / (float)DD;
  __syncthreads();
  float v = 0.f;
#pragma unroll
  for (int i = 0; i < DD / 256; ++i) { const float dl = loc[i] - mu; v += dl * dl; }
  red[t] = v; __syncthreads();
  for (int k = 128; k > 0; k >>= 1) { if (t < k) red[t] += red[t + k]; __syncthreads(); }
  const float rs = rsqrtf(red[0] / (float)DD + LN_EPS);
#pragma unroll
  for (int i = 0; i < DD / 256; ++i) {
    const int d = t + i * 256;
    advn[base + d] = (loc[i] - mu) * rs * ln_w[d] + ln_b[d];
  }
}

// ---------------------------------------------------------------------------
// Causal row softmax in place on logits; accumulates probs/NH into attn_w.
// Grid: (L, B). One block per row.
// ---------------------------------------------------------------------------
__global__ __launch_bounds__(256) void softmax_kernel(float* __restrict__ logits,
                                                      float* __restrict__ attn_w)
{
  __shared__ float red[256];
  const int l = blockIdx.x, b = blockIdx.y, t = threadIdx.x;
  float* row = logits + ((long)b * LL + l) * LL;
  const int n = l + 1;

  float mx = -INFINITY;
  for (int c = t; c < n; c += 256) mx = fmaxf(mx, row[c]);
  red[t] = mx; __syncthreads();
  for (int k = 128; k > 0; k >>= 1) { if (t < k) red[t] = fmaxf(red[t], red[t + k]); __syncthreads(); }
  mx = red[0]; __syncthreads();

  float s = 0.f;
  for (int c = t; c < n; c += 256) { const float e = expf(row[c] - mx); row[c] = e; s += e; }
  red[t] = s; __syncthreads();
  for (int k = 128; k > 0; k >>= 1) { if (t < k) red[t] += red[t + k]; __syncthreads(); }
  const float inv = 1.f / red[0];

  float* aw = attn_w + ((long)b * LL + l) * LL;
  for (int c = t; c < n; c += 256) {
    const float p = row[c] * inv;
    row[c] = p;
    aw[c] += p * (1.f / (float)NHH);
  }
  for (int c = n + t; c < LL; c += 256) row[c] = 0.f;  // kill stale upper triangle
}

__global__ void zero_kernel(float* __restrict__ p, long n)
{
  const long i = (long)blockIdx.x * blockDim.x + threadIdx.x;
  if (i < n) p[i] = 0.f;
}

__global__ void combine_kernel(const float* __restrict__ z,
                               const float* __restrict__ aout,
                               const float* __restrict__ mlp,
                               float* __restrict__ zn)
{
  const long i = (long)blockIdx.x * blockDim.x + threadIdx.x;
  if (i < (long)BB * LL * DD)
    zn[i] = z[i] + ALPHA_ATTN * aout[i] + ALPHA_MLP * mlp[i];
}

// ---------------------------------------------------------------------------
extern "C" void kernel_launch(void* const* d_in, const int* in_sizes, int n_in,
                              void* d_out, int out_size, void* d_ws, size_t ws_size,
                              hipStream_t stream)
{
  const float* z     = (const float*)d_in[0];
  const float* Wq    = (const float*)d_in[1];
  const float* bq    = (const float*)d_in[2];
  const float* Wk    = (const float*)d_in[3];
  const float* bk    = (const float*)d_in[4];
  const float* Wv    = (const float*)d_in[5];
  const float* bv    = (const float*)d_in[6];
  const float* Wcoh  = (const float*)d_in[7];
  const float* gamma = (const float*)d_in[8];
  const float* rg_w1 = (const float*)d_in[9];
  const float* rg_b1 = (const float*)d_in[10];
  const float* rg_w2 = (const float*)d_in[11];
  const float* rg_b2 = (const float*)d_in[12];
  const float* nu_df = (const float*)d_in[13];
  const float* nu_ad = (const float*)d_in[14];
  const float* nu_w1 = (const float*)d_in[15];
  const float* nu_b1 = (const float*)d_in[16];
  const float* nu_w2 = (const float*)d_in[17];
  const float* nu_b2 = (const float*)d_in[18];
  const float* Wu    = (const float*)d_in[19];
  const float* bu    = (const float*)d_in[20];
  const float* Wg    = (const float*)d_in[21];
  const float* bg    = (const float*)d_in[22];
  const float* ln_w  = (const float*)d_in[23];
  const float* ln_b  = (const float*)d_in[24];
  const float* Cw    = (const float*)d_in[25];
  const float* Cb    = (const float*)d_in[26];

  const long M4  = (long)BB * LL * DD;   // 4,194,304 floats
  const long MLL = (long)BB * LL * LL;   // 8,388,608 floats
  float* ws = (float*)d_ws;
  // Workspace layout (slot reuse over the dependency timeline): ~201 MB total.
  float* pool = ws;                 // later: adv_n
  float* Qb   = ws + 1 * M4;        // later: U
  float* Kb   = ws + 2 * M4;        // later: G
  float* Vb   = ws + 3 * M4;
  float* Qr   = ws + 4 * M4;
  float* Kr   = ws + 5 * M4;
  float* gk   = ws + 6 * M4;        // later: z_glu, later: mlp_out
  float* aout = ws + 7 * M4;
  float* gb   = ws + 8 * M4;        // B*L*L
  float* logit = ws + 8 * M4 + MLL; // B*L*L (probs in place)
  float* gatev = ws + 8 * M4 + 2 * MLL;
  float* nubv  = gatev + (long)BB * LL;
  float* advn = pool;
  float* zglu = gk;
  float* mlpo = gk;
  float* Ub = Qb; float* Gb = Kb;

  float* zn     = (float*)d_out;
  float* attn_w = zn + M4;

  const dim3 blk(256);
  const dim3 gProj((DD + 127) / 128, (BB * LL + 127) / 128, 1); // [4096 x 1024]
  const dim3 gSq((LL + 127) / 128, (LL + 127) / 128, BB);       // [2048 x 2048] x B
  const dim3 gAO(1, (LL + 127) / 128, BB);                      // [2048 x 64]   x B

  // 1. causal cumulative-mean pool
  pool_kernel<<<(BB * DD + 255) / 256, blk, 0, stream>>>(z, pool);

  // 2. gate + nu_b (small MLPs on pool)
  gate_nu_kernel<<<BB * LL, blk, 0, stream>>>(pool, rg_w1, rg_b1, rg_w2, rg_b2,
                                              nu_w1, nu_b1, nu_w2, nu_b2,
                                              nu_df, nu_ad, gatev, nubv);

  // 3. QKV + coherence projections (WMMA)
  wmma_gemm_kernel<<<gProj, blk, 0, stream>>>(z, DD, 0, Wq, DD, 0, Qb, DD, 0,
      bq, nullptr, 0, 0, nullptr, 1.0f, BB * LL, DD, DD, 0);
  wmma_gemm_kernel<<<gProj, blk, 0, stream>>>(z, DD, 0, Wk, DD, 0, Kb, DD, 0,
      bk, nullptr, 0, 0, nullptr, 1.0f, BB * LL, DD, DD, 0);
  wmma_gemm_kernel<<<gProj, blk, 0, stream>>>(z, DD, 0, Wv, DD, 0, Vb, DD, 0,
      bv, nullptr, 0, 0, nullptr, 1.0f, BB * LL, DD, DD, 0);
  wmma_gemm_kernel<<<gProj, blk, 0, stream>>>(pool, DD, 0, Wcoh, DD, 0, gk, DD, 0,
      nullptr, nullptr, 0, 0, nullptr, 1.0f, BB * LL, DD, DD, 0);

  // 4. gb[b] = tanh(gamma) * D^-0.5 * (gk[b] @ K_raw[b]^T)  (WMMA, nt)
  wmma_gemm_kernel<<<gSq, blk, 0, stream>>>(gk, DD, (long)LL * DD,
      Kb, DD, (long)LL * DD, gb, LL, (long)LL * LL,
      nullptr, nullptr, 0, 0, gamma, 0.03125f, LL, LL, DD, 1);

  // 5. RoPE (gated fast/slow on Q,K; fixed GLU base on z)  [gk slot -> z_glu]
  rope_kernel<<<(unsigned)((M4 + 255) / 256), blk, 0, stream>>>(Qb, Kb, z, gatev,
                                                                Qr, Kr, zglu);

  // 6. GLU branch GEMMs (WMMA), reusing Q/K slots for U/G
  wmma_gemm_kernel<<<gProj, blk, 0, stream>>>(zglu, DD, 0, Wu, DD, 0, Ub, DD, 0,
      bu, nullptr, 0, 0, nullptr, 1.0f, BB * LL, DD, DD, 0);
  wmma_gemm_kernel<<<gProj, blk, 0, stream>>>(zglu, DD, 0, Wg, DD, 0, Gb, DD, 0,
      bg, nullptr, 0, 0, nullptr, 1.0f, BB * LL, DD, DD, 0);

  // 7. adv + layernorm  [pool slot -> adv_n]
  glu_ln_kernel<<<BB * LL, blk, 0, stream>>>(Ub, Gb, nubv, ln_w, ln_b, advn);

  // 8. mlp_out = adv_n @ Cw + Cb  (WMMA)  [z_glu slot -> mlp_out]
  wmma_gemm_kernel<<<gProj, blk, 0, stream>>>(advn, DD, 0, Cw, DD, 0, mlpo, DD, 0,
      Cb, nullptr, 0, 0, nullptr, 1.0f, BB * LL, DD, DD, 0);

  // 9. attention: zero attn_w, then loop heads: logits -> softmax -> P@V
  zero_kernel<<<(unsigned)((MLL + 255) / 256), blk, 0, stream>>>(attn_w, MLL);
  for (int h = 0; h < NHH; ++h) {
    // logits[b] = 0.125 * Qh[b] @ Kh[b]^T + gb[b]
    wmma_gemm_kernel<<<gSq, blk, 0, stream>>>(Qr + h * HDD, DD, (long)LL * DD,
        Kr + h * HDD, DD, (long)LL * DD, logit, LL, (long)LL * LL,
        nullptr, gb, LL, (long)LL * LL, nullptr, 0.125f, LL, LL, HDD, 1);
    // causal softmax (in place) + accumulate mean-over-heads into attn_w
    softmax_kernel<<<dim3(LL, BB), blk, 0, stream>>>(logit, attn_w);
    // attn_out[:, h*64:(h+1)*64] = probs @ Vh
    wmma_gemm_kernel<<<gAO, blk, 0, stream>>>(logit, LL, (long)LL * LL,
        Vb + h * HDD, DD, (long)LL * DD, aout + h * HDD, DD, (long)LL * DD,
        nullptr, nullptr, 0, 0, nullptr, 1.0f, LL, HDD, LL, 0);
  }

  // 10. z_next = z + 0.42 * attn_out + 1.07 * mlp_out
  combine_kernel<<<(unsigned)((M4 + 255) / 256), blk, 0, stream>>>(z, aout, mlpo, zn);
}
